// BGU_76828374991063
// MI455X (gfx1250) — matrix operand order repaired and checked by reading, not compiled, not run on hardware
//
#include <hip/hip_runtime.h>
#include <hip/hip_bf16.h>

typedef __attribute__((ext_vector_type(16))) _Float16 v16h;
typedef __attribute__((ext_vector_type(8)))  _Float16 v8h;
typedef __attribute__((ext_vector_type(8)))  float    v8f;
typedef __attribute__((ext_vector_type(4)))  unsigned int v4u;
typedef __attribute__((ext_vector_type(4)))  int      v4i;
typedef __attribute__((ext_vector_type(8)))  int      v8i;

#define HH 512
#define WW 512
#define GH 32
#define GW 32
#define GD 16
#define NB 2
#define GCELLS (GH*GW*GD)
#define LAMBDA 1e-7f
#define NXMAX 64   // max pixel-support width of one grid column (~36)

// ---------------------------------------------------------------------------
// Kernel 0: zero the 12-float global-sum scratch.
// ---------------------------------------------------------------------------
__global__ void bgu_zero_sums(float* s) {
    if (threadIdx.x < 12) s[threadIdx.x] = 0.f;
}

// ---------------------------------------------------------------------------
// Kernel 1: per-channel plane sums (trilinear weights sum to 1, so the
// reference's S_sum/T_sum/counts.sum collapse to plain image sums).
// sums[0..5] = sum(input planes), sums[6..11] = sum(output planes).
// ---------------------------------------------------------------------------
__global__ __launch_bounds__(256) void bgu_plane_sum(
    const float* __restrict__ inp, const float* __restrict__ outp,
    float* __restrict__ sums)
{
    const int p = blockIdx.y;                       // 0..11
    const float* src = (p < 6) ? (inp  + (size_t)p      * HH * WW)
                               : (outp + (size_t)(p - 6) * HH * WW);
    float s = 0.f;
    for (int i = blockIdx.x * blockDim.x + threadIdx.x; i < HH * WW;
         i += gridDim.x * blockDim.x)
        s += src[i];
    #pragma unroll
    for (int off = 16; off > 0; off >>= 1) s += __shfl_down(s, off, 32);
    __shared__ float red[8];
    const int wid = threadIdx.x >> 5;
    if ((threadIdx.x & 31) == 0) red[wid] = s;
    __syncthreads();
    if (threadIdx.x == 0) {
        float t = 0.f;
        #pragma unroll
        for (int w = 0; w < 8; ++w) t += red[w];
        atomicAdd(&sums[p], t);
    }
}

// ---------------------------------------------------------------------------
// TDM: issue a 2D tile load (tile_h rows x tile_w f32, row stride in
// elements) from global into LDS at byte offset lds_off.  D# per CDNA5 ISA
// ch.8 (group0: count|lds|gaddr|type=2 ; group1: data_size=4B, dims, stride).
// ---------------------------------------------------------------------------
__device__ __forceinline__ void tdm_load_2d(unsigned lds_off, const float* gptr,
                                            unsigned tile_w, unsigned tile_h,
                                            unsigned long long stride_elems)
{
#if defined(__HIP_DEVICE_COMPILE__)
    const unsigned long long ga = (unsigned long long)(uintptr_t)gptr;
    v4u g0;
    g0[0] = 1u;                                        // count=1 (valid user D#)
    g0[1] = lds_off;                                   // lds_addr (bytes)
    g0[2] = (unsigned)(ga & 0xFFFFFFFFu);              // global_addr[31:0]
    g0[3] = (unsigned)((ga >> 32) & 0x01FFFFFFu)       // global_addr[56:32]
          | 0x80000000u;                               // type = 2 ("image")
    v8i g1;
    g1[0] = (int)(2u << 16);                           // data_size = 4 bytes
    g1[1] = (int)((tile_w & 0xFFFFu) << 16);           // tensor_dim0[15:0]
    g1[2] = (int)(((tile_w >> 16) & 0xFFFFu)           // tensor_dim0[31:16]
          | ((tile_h & 0xFFFFu) << 16));               // tensor_dim1[15:0]
    g1[3] = (int)(((tile_h >> 16) & 0xFFFFu)           // tensor_dim1[31:16]
          | ((tile_w & 0xFFFFu) << 16));               // tile_dim0
    g1[4] = (int)(tile_h & 0xFFFFu);                   // tile_dim1 (tile_dim2=0)
    g1[5] = (int)(stride_elems & 0xFFFFFFFFu);         // dim0_stride[31:0]
    g1[6] = (int)((stride_elems >> 32) & 0xFFFFu);     // dim0_stride[47:32]
    g1[7] = 0;                                         // dim1_stride = 0 (2D)
    const v4i gz = {0, 0, 0, 0};
#if __has_include(<hip/amd_detail/amd_gfx1250_TDM.h>)
    const v8i gz8 = {0, 0, 0, 0, 0, 0, 0, 0};
    __builtin_amdgcn_tensor_load_to_lds(g0, g1, gz, gz, gz8, 0);
#else
    __builtin_amdgcn_tensor_load_to_lds(g0, g1, gz, gz, 0);
#endif
#else
    (void)lds_off; (void)gptr; (void)tile_w; (void)tile_h; (void)stride_elems;
#endif
}

__device__ __forceinline__ unsigned lds_offset_of(const void* p) {
    // generic pointer to __shared__: low 32 bits are the wave-relative LDS addr
    return (unsigned)(unsigned long long)(uintptr_t)p;
}

// ---------------------------------------------------------------------------
// Kernel 2: one 512-thread workgroup (16 waves) per (b, cy, cx) grid column;
// wave z owns cell (cy,cx,z).  Pixel rows are DMA'd once by the Tensor Data
// Mover into double-buffered LDS staging: wave 0 issues row py+1 into the
// alternate buffer, then s_wait_tensorcnt(3) -- TDM ops from one wave retire
// in order, so row py's three loads are complete while row py+1's stay in
// flight behind the compute.  Each wave WMMA-accumulates its 7x7 moment
// matrix M_z = sum_p w_p(z) u_p u_p^T (S = M[0:4,0:4], T = M[4:7,0:4]) with
// v_wmma_f32_16x16x32_f16 (K = 32 pixels/instruction); lane 0 of each wave
// then regularizes and solves gamma = T_reg * S_reg^{-1} in-register.
// ---------------------------------------------------------------------------
__global__ __launch_bounds__(512) void bgu_cell_wmma(
    const float* __restrict__ inp,    // [B,3,H,W]
    const float* __restrict__ guide,  // [B,H,W]
    const float* __restrict__ outp,   // [B,3,H,W]
    const float* __restrict__ sums,   // [12]
    float* __restrict__ gamma_out)    // [B,G,3,4] flat
{
    __shared__ float    tdmIn [2][3 * NXMAX];  // TDM staging: 3 input rows
    __shared__ float    tdmOut[2][3 * NXMAX];  // TDM staging: 3 output rows
    __shared__ float    tdmG  [2][NXMAX];      // TDM staging: guide row
    __shared__ _Float16 uuT[16][32];           // B staging: u[feature][pixel]
    __shared__ _Float16 wuT[16][16][32];       // A staging per z: (w*u)[f][px]
    __shared__ float    gwxy[32];              // per-pixel wtx*wty
    __shared__ float    gfz[32];               // per-pixel z fraction
    __shared__ int      gz0[32];               // per-pixel floor z
    __shared__ float    stageS[16][28];        // per-wave solve staging

    const int tid  = threadIdx.x;
    const int lane = tid & 31;
    const int wv   = tid >> 5;                 // wave id == cz

    const int blk = blockIdx.x;
    const int b   = blk / (GH * GW);
    const int rr  = blk - b * (GH * GW);
    const int cy  = rr / GW;
    const int cx  = rr - cy * GW;

    // one-time LDS init: uuT rows 3 (=1) and 7..15 (=0); wuT feature rows 8..15
    for (int i = tid; i < 16 * 32; i += 512) {
        const int mm = i >> 5, kk = i & 31;
        if (mm == 3)      uuT[mm][kk] = (_Float16)1.f;
        else if (mm >= 7) uuT[mm][kk] = (_Float16)0.f;
    }
    for (int i = tid; i < 16 * 8 * 32; i += 512) {
        const int zz = i >> 8, mm = 8 + ((i >> 5) & 7), kk = i & 31;
        wuT[zz][mm][kk] = (_Float16)0.f;
    }
    __syncthreads();

    // pixel support of this (cy,cx) column (grid pitch 512/31 px, + slack;
    // the exact weights below zero out over-included pixels)
    const float pitch = 512.f / 31.f;
    int px_lo = (int)floorf((cx - 1) * pitch - 0.5f); if (px_lo < 0)   px_lo = 0;
    int px_hi = (int)ceilf ((cx + 1) * pitch - 0.5f); if (px_hi > 511) px_hi = 511;
    int py_lo = (int)floorf((cy - 1) * pitch - 0.5f); if (py_lo < 0)   py_lo = 0;
    int py_hi = (int)ceilf ((cy + 1) * pitch - 0.5f); if (py_hi > 511) py_hi = 511;
    const int nx = px_hi - px_lo + 1;          // <= ~36 < NXMAX

    const size_t plane = (size_t)HH * WW;
    const float* in0 = inp   + (size_t)b * 3 * plane;
    const float* ou0 = outp  + (size_t)b * 3 * plane;
    const float* gu0 = guide + (size_t)b * plane;

    unsigned offIn[2], offOut[2], offG[2];
    #pragma unroll
    for (int q = 0; q < 2; ++q) {
        offIn[q]  = lds_offset_of(&tdmIn[q][0]);
        offOut[q] = lds_offset_of(&tdmOut[q][0]);
        offG[q]   = lds_offset_of(&tdmG[q][0]);
    }

    v8f acc = {};
    const float sgrid = 31.f / 512.f;

    // prologue: DMA first row into buffer 0
    if (wv == 0) {
        const int rowoff = py_lo * WW + px_lo;
        tdm_load_2d(offIn[0],  in0 + rowoff, (unsigned)nx, 3u, (unsigned long long)plane);
        tdm_load_2d(offOut[0], ou0 + rowoff, (unsigned)nx, 3u, (unsigned long long)plane);
        tdm_load_2d(offG[0],   gu0 + rowoff, (unsigned)nx, 1u, (unsigned long long)WW);
    }

    for (int py = py_lo; py <= py_hi; ++py) {
        const int cur = (py - py_lo) & 1;
        __syncthreads();   // all waves done reading buffer cur (one row ago)
        if (wv == 0) {
            if (py < py_hi) {   // pipeline: start row py+1, wait only for row py
                const int rowoff = (py + 1) * WW + px_lo;
                const int nxt = cur ^ 1;
                tdm_load_2d(offIn[nxt],  in0 + rowoff, (unsigned)nx, 3u, (unsigned long long)plane);
                tdm_load_2d(offOut[nxt], ou0 + rowoff, (unsigned)nx, 3u, (unsigned long long)plane);
                tdm_load_2d(offG[nxt],   gu0 + rowoff, (unsigned)nx, 1u, (unsigned long long)WW);
#if defined(__HIP_DEVICE_COMPILE__)
                __builtin_amdgcn_s_wait_tensorcnt(3);  // row py landed; py+1 in flight
#endif
            } else {
#if defined(__HIP_DEVICE_COMPILE__)
                __builtin_amdgcn_s_wait_tensorcnt(0);  // drain for the last row
#endif
            }
        }
        __syncthreads();   // staging buffer `cur` visible to all waves

        const float* curIn  = tdmIn[cur];
        const float* curOut = tdmOut[cur];
        const float* curG   = tdmG[cur];

        const float gy  = (py + 0.5f) * sgrid;
        const int   y0  = (int)floorf(gy);
        const float fy  = gy - (float)y0;
        const float wty = (y0 == cy) ? (1.f - fy) : ((y0 + 1 == cy) ? fy : 0.f);

        for (int xx = px_lo; xx <= px_hi; xx += 32) {
            const int kkrow  = xx - px_lo + lane;      // pixel index within row
            const int kk     = kkrow >= nx ? nx - 1 : kkrow;
            const bool valid = kkrow < nx;

            // ---- phase A: shared staging (waves 0..6) -----------------
            if (wv < 3) {                              // input channels 0..2
                uuT[wv][lane] = (_Float16)curIn[wv * nx + kk];
            } else if (wv == 3) {                      // xy weights + z info
                const int   px  = px_lo + kk;
                const float gx  = (px + 0.5f) * sgrid;
                const int   x0  = (int)floorf(gx);
                const float fx  = gx - (float)x0;
                const float wtx = (x0 == cx) ? (1.f - fx)
                                             : ((x0 + 1 == cx) ? fx : 0.f);
                const float gz = curG[kk] * 15.f;
                int z0 = (int)floorf(gz);
                z0 = z0 < 0 ? 0 : (z0 > 15 ? 15 : z0);
                gwxy[lane] = valid ? (wtx * wty) : 0.f;
                gz0[lane]  = z0;
                gfz[lane]  = gz - (float)z0;
            } else if (wv < 7) {                       // output channels 0..2
                uuT[wv][lane] = (_Float16)curOut[(wv - 4) * nx + kk];
            }
            __syncthreads();

            // ---- phase B: per-z weighted A staging (all 512 threads) --
            {
                const int z0p = gz0[lane];
                const int z1p = (z0p + 1 > 15) ? 15 : z0p + 1;
                const float fz = gfz[lane];
                const float wz = ((z0p == wv) ? (1.f - fz) : 0.f)
                               + ((z1p == wv) ? fz : 0.f);
                const _Float16 wh = (_Float16)(gwxy[lane] * wz);
                #pragma unroll
                for (int m = 0; m < 8; ++m)
                    wuT[wv][m][lane] = wh * uuT[m][lane];
            }
            __syncthreads();

            // ---- phase C: fragments + WMMA (each wave, its own z) -----
            // A (16-bit 16x32): lane L -> M = L&15;
            // lanes 0-15: K=0..7,16..23 ; lanes 16-31: K=8..15,24..31
            const int m      = lane & 15;
            const int kbaseA = (lane >> 4) << 3;
            const v8h alo = *(const v8h*)&wuT[wv][m][kbaseA];
            const v8h ahi = *(const v8h*)&wuT[wv][m][kbaseA + 16];
            v16h a;
            #pragma unroll
            for (int i = 0; i < 8; ++i) { a[i] = alo[i]; a[8 + i] = ahi[i]; }

            // B (16-bit 32x16): lane L -> N = L&15;
            // lanes 0-15: K=0..7,16..23 ; lanes 16-31: K=8..15,24..31
            const int kbaseB = (lane >> 4) << 3;
            const v8h blo = *(const v8h*)&uuT[m][kbaseB];
            const v8h bhi = *(const v8h*)&uuT[m][kbaseB + 16];
            v16h bm;
            #pragma unroll
            for (int i = 0; i < 8; ++i) { bm[i] = blo[i]; bm[8 + i] = bhi[i]; }

            acc = __builtin_amdgcn_wmma_f32_16x16x32_f16(
                false, a, false, bm, (short)0, acc, false, false);

            __syncthreads();   // protect uuT/wuT WAR before next chunk
        }
    }

    // D layout: lane n (0..15) holds column n, acc[m] = D[m][n], m = 0..7.
    if (lane < 4) {
        #pragma unroll
        for (int mm = 0; mm < 7; ++mm) stageS[wv][mm * 4 + lane] = acc[mm];
    }
    __syncthreads();

    if (lane == 0) {
        float S[4][4], T[3][4];
        #pragma unroll
        for (int i = 0; i < 4; ++i)
            #pragma unroll
            for (int j = 0; j < 4; ++j) S[i][j] = stageS[wv][i * 4 + j];
        #pragma unroll
        for (int k = 0; k < 3; ++k)
            #pragma unroll
            for (int j = 0; j < 4; ++j) T[k][j] = stageS[wv][(4 + k) * 4 + j];

        const float counts = S[3][3];
        const float wl  = LAMBDA * (counts + 1.f);
        const float wlg = LAMBDA * ((float)(HH * WW) + 1.f);
        const bool  empty = (counts == 0.f);

        float mg[3];
        #pragma unroll
        for (int k = 0; k < 3; ++k) {
            const float gglob = sums[6 + b * 3 + k] / (sums[b * 3 + k] + wlg);
            const float gloc  = T[k][3] / (S[k][3] + wl);
            mg[k] = empty ? gglob : gloc;
        }

        // Gauss-Jordan inverse of S_reg (SPD, no pivoting needed)
        float M[4][8];
        #pragma unroll
        for (int i = 0; i < 4; ++i)
            #pragma unroll
            for (int j = 0; j < 4; ++j) {
                M[i][j]     = S[i][j] + (i == j ? wl : 0.f);
                M[i][4 + j] = (i == j) ? 1.f : 0.f;
            }
        #pragma unroll
        for (int c2 = 0; c2 < 4; ++c2) {
            const float ip = 1.f / M[c2][c2];
            #pragma unroll
            for (int j = 0; j < 8; ++j) M[c2][j] *= ip;
            #pragma unroll
            for (int r2 = 0; r2 < 4; ++r2) {
                if (r2 == c2) continue;
                const float f = M[r2][c2];
                #pragma unroll
                for (int j = 0; j < 8; ++j) M[r2][j] -= f * M[c2][j];
            }
        }

        const int cell = b * GCELLS + cy * (GW * GD) + cx * GD + wv;
        float* o = gamma_out + (size_t)cell * 12;
        #pragma unroll
        for (int k = 0; k < 3; ++k) {
            #pragma unroll
            for (int j = 0; j < 4; ++j) {
                float g = 0.f;
                #pragma unroll
                for (int i = 0; i < 4; ++i) {
                    const float treg = T[k][i] + ((i == k) ? wl * mg[k] : 0.f);
                    g += treg * M[i][4 + j];
                }
                o[k * 4 + j] = g;
            }
        }
    }
}

// ---------------------------------------------------------------------------
extern "C" void kernel_launch(void* const* d_in, const int* in_sizes, int n_in,
                              void* d_out, int out_size, void* d_ws, size_t ws_size,
                              hipStream_t stream) {
    (void)in_sizes; (void)n_in; (void)out_size; (void)ws_size;
    const float* inp   = (const float*)d_in[0];  // [2,3,512,512]
    const float* guide = (const float*)d_in[1];  // [2,512,512]
    const float* outp  = (const float*)d_in[2];  // [2,3,512,512]
    float* sums = (float*)d_ws;                  // 12 floats
    float* out  = (float*)d_out;                 // [2,32,32,16,3,4]

    bgu_zero_sums<<<1, 32, 0, stream>>>(sums);
    bgu_plane_sum<<<dim3(32, 12), 256, 0, stream>>>(inp, outp, sums);
    bgu_cell_wmma<<<NB * GH * GW, 512, 0, stream>>>(inp, guide, outp, sums, out);
}
